// GraphAttentionLayer_70274254897801
// MI455X (gfx1250) — matrix operand order repaired and checked by reading, not compiled, not run on hardware
//
#include <hip/hip_runtime.h>
#include <hip/hip_bf16.h>
#include <math.h>

// ---------------------------------------------------------------------------
// GAT layer for MI455X (gfx1250, wave32, WMMA), flash-style single pass.
//   N = 8192, F_IN = 512, F_OUT = 128
// bf16 WMMA (16x16x32, f32 accum) for both GEMMs; fp32 online softmax fused
// into the attention matmul so adj (256MB) is streamed exactly ONCE
// (~11.5us @ 23.3TB/s = roofline floor). h stored transposed so B-fragments
// are contiguous ds_load_b128; h tiles staged with CDNA5 async global->LDS
// copies (ASYNCcnt / s_wait_asynccnt).
// ---------------------------------------------------------------------------

#define GAT_N     8192
#define GAT_FIN   512
#define GAT_FOUT  128

#define GAT_USE_ASYNC 1

typedef __attribute__((ext_vector_type(16))) __bf16 v16bf;
typedef __attribute__((ext_vector_type(8)))  float  v8f;

__device__ __forceinline__ float gat_lrelu(float x) {
    return x >= 0.0f ? x : 0.2f * x;
}

#if GAT_USE_ASYNC
// One per-lane 16B async copy: global -> LDS, tracked by ASYNCcnt.
// LDS operand = wave-relative byte address (low 32 bits of the flat pointer).
__device__ __forceinline__ void gat_async_b128(const __bf16* g, __bf16* l) {
    asm volatile("global_load_async_to_lds_b128 %0, %1, off"
                 :
                 : "v"((uint32_t)(uintptr_t)l), "v"((uint64_t)(uintptr_t)g)
                 : "memory");
}
__device__ __forceinline__ void gat_wait_async() {
    asm volatile("s_wait_asynccnt 0x0" ::: "memory");
}
#endif

// ---------------------------------------------------------------------------
// Kernel 1: h = x @ W (fp32 in -> bf16 out, row-major h + transposed h_t).
// Block = 16 rows x 128 cols, 8 waves, K steps of 32; W tile staged
// K-contiguous so both fragments load as ds_load_b128.
// ---------------------------------------------------------------------------
__global__ void __launch_bounds__(256)
gat_k1_gemm_hW(const float* __restrict__ x, const float* __restrict__ W,
               __bf16* __restrict__ h_bf, __bf16* __restrict__ h_t) {
    __shared__ __align__(16) __bf16 xs[16][40];
    __shared__ __align__(16) __bf16 wt[128][40];

    const int t    = threadIdx.x;
    const int w    = t >> 5;
    const int lane = t & 31;
    const int row0 = blockIdx.x * 16;

    const int m    = lane & 15;
    const int kbA  = (lane >> 4) * 8;
    const int kbB  = (lane >> 4) * 16;
    const int n    = lane & 15;
    const int col  = 16 * w + n;

    v8f acc = {};

    for (int k0 = 0; k0 < GAT_FIN; k0 += 32) {
        {
            int i = t >> 5, kk = t & 31;
            xs[i][kk]     = (__bf16)x[(size_t)(row0 + i) * GAT_FIN + k0 + kk];
            xs[i + 8][kk] = (__bf16)x[(size_t)(row0 + i + 8) * GAT_FIN + k0 + kk];
        }
        #pragma unroll
        for (int q = 0; q < 16; ++q) {
            int e = t + 256 * q;
            int i = e >> 7, nn = e & 127;
            wt[nn][i] = (__bf16)W[(size_t)(k0 + i) * GAT_FOUT + nn];
        }
        __syncthreads();

        v16bf afrag, bfrag;
        #pragma unroll
        for (int e = 0; e < 16; ++e) {
            int ka = kbB + e;
            afrag[e] = xs[m][ka];
            bfrag[e] = wt[col][kbB + e];
        }
        acc = __builtin_amdgcn_wmma_f32_16x16x32_bf16(
            false, afrag, false, bfrag, (short)0, acc, false, false);
        __syncthreads();
    }

    const int mh = 8 * (lane >> 4);
    #pragma unroll
    for (int r = 0; r < 8; ++r) {
        __bf16 v = (__bf16)acc[r];
        h_bf[(size_t)(row0 + mh + r) * GAT_FOUT + col] = v;
        h_t[(size_t)col * GAT_N + row0 + mh + r]       = v;
    }
}

// ---------------------------------------------------------------------------
// Kernel 2: s_src[i] = h[i,:] . a[0:128],  s_dst[i] = h[i,:] . a[128:256]
// ---------------------------------------------------------------------------
__global__ void __launch_bounds__(128)
gat_k2_logits(const __bf16* __restrict__ h_bf, const float* __restrict__ a,
              float* __restrict__ s_src, float* __restrict__ s_dst) {
    __shared__ float r1[128];
    __shared__ float r2[128];
    const int row = blockIdx.x;
    const int t   = threadIdx.x;

    float v = (float)h_bf[(size_t)row * GAT_FOUT + t];
    r1[t] = v * a[t];
    r2[t] = v * a[GAT_FOUT + t];
    __syncthreads();
    #pragma unroll
    for (int off = 64; off > 0; off >>= 1) {
        if (t < off) { r1[t] += r1[t + off]; r2[t] += r2[t + off]; }
        __syncthreads();
    }
    if (t == 0) { s_src[row] = r1[0]; s_dst[row] = r2[0]; }
}

// ---------------------------------------------------------------------------
// Kernel 3 (fused): out = elu( softmax_row(mask(adj, lrelu(si+sj))) @ h )
// Flash-style online softmax: single streaming pass over adj.
// Block = 16 rows x 128 cols (8 waves), K chunks of 64 (2 WMMAs / chunk).
// P is kept UNNORMALIZED (exp(sc - running_max)); the f32 accumulator is
// rescaled by exp(m_old - m_new) per row; division by the row sum at the end.
// ---------------------------------------------------------------------------
__global__ void __launch_bounds__(256)
gat_k3_fused_attn(const float* __restrict__ adj, const __bf16* __restrict__ h_t,
                  const float* __restrict__ s_src, const float* __restrict__ s_dst,
                  float* __restrict__ out) {
    __shared__ __align__(16) __bf16 p_lds[16][72];   // P tile rows (K-contig)
    __shared__ __align__(16) __bf16 ht[128][72];     // h tile cols (K-contig)
    __shared__ float red[16][17];                    // padded: conflict-free
    __shared__ float rm[16], rs[16], rscale[16], ssrc[16];

    const int t    = threadIdx.x;
    const int w    = t >> 5;
    const int lane = t & 31;
    const int row0 = blockIdx.x * 16;

    if (t < 16) {
        rm[t] = -3.0e38f;   // finite sentinel -> NaN-free rescale factors
        rs[t] = 0.0f;
        rscale[t] = 1.0f;
        ssrc[t] = s_src[row0 + t];
    }
    __syncthreads();

    const int m    = lane & 15;
    const int kbA  = (lane >> 4) * 8;
    const int kbB  = (lane >> 4) * 16;
    const int n    = lane & 15;
    const int col  = 16 * w + n;
    const int mh   = 8 * (lane >> 4);

    const int ti   = t >> 4;      // P row owned by this thread (0..15)
    const int slot = t & 15;      // 16 threads per row
    const int jb   = 4 * slot;    // 4-elem slice within the 64-wide chunk

    v8f acc = {};

    for (int j0 = 0; j0 < GAT_N; j0 += 64) {
        // ---- phase A: async h stage + adj slice + scores + local max ----
        {
            int nn = t >> 1, half = t & 1;           // 64B per thread
            const __bf16* g = h_t + (size_t)nn * GAT_N + j0 + 32 * half;
            __bf16* l = &ht[nn][32 * half];
#if GAT_USE_ASYNC
            #pragma unroll
            for (int q = 0; q < 4; ++q) gat_async_b128(g + 8 * q, l + 8 * q);
#else
            #pragma unroll
            for (int q = 0; q < 4; ++q)
                ((uint4*)l)[q] = ((const uint4*)g)[q];
#endif
        }
        const float4 av = *(const float4*)&adj[(size_t)(row0 + ti) * GAT_N + j0 + jb];
        const float4 sd = *(const float4*)&s_dst[j0 + jb];
        const float si  = ssrc[ti];
        // masked entries -> -inf: exp(sc - m) == 0 for any finite running max
        float sc0 = av.x > 0.0f ? gat_lrelu(si + sd.x) : -INFINITY;
        float sc1 = av.y > 0.0f ? gat_lrelu(si + sd.y) : -INFINITY;
        float sc2 = av.z > 0.0f ? gat_lrelu(si + sd.z) : -INFINITY;
        float sc3 = av.w > 0.0f ? gat_lrelu(si + sd.w) : -INFINITY;
        red[ti][slot] = fmaxf(fmaxf(sc0, sc1), fmaxf(sc2, sc3));
        if (j0 + 64 < GAT_N && t < 16) {
            __builtin_prefetch(&adj[(size_t)(row0 + t) * GAT_N + j0 + 64], 0, 1);
        }
        __syncthreads();                             // B1: red (max) ready

        // ---- phase B: row owners merge running max, emit rescale ----
        if (t < 16) {
            float M = rm[t];
            #pragma unroll
            for (int s2 = 0; s2 < 16; ++s2) M = fmaxf(M, red[t][s2]);
            rscale[t] = __expf(rm[t] - M);   // 1 if unchanged; 0 if first hit
            rm[t] = M;
        }
        __syncthreads();                             // B2: rm/rscale ready

        // ---- phase C: p values (unnormalized), packed store, local sum,
        //               accumulator rescale ----
        const float Mi = rm[ti];
        float p0 = __expf(sc0 - Mi);
        float p1 = __expf(sc1 - Mi);
        float p2 = __expf(sc2 - Mi);
        float p3 = __expf(sc3 - Mi);
        union { __bf16 b[4]; uint2 u; } pk;
        pk.b[0] = (__bf16)p0; pk.b[1] = (__bf16)p1;
        pk.b[2] = (__bf16)p2; pk.b[3] = (__bf16)p3;
        *(uint2*)&p_lds[ti][jb] = pk.u;              // ds_store_b64
        red[ti][slot] = p0 + p1 + p2 + p3;
        #pragma unroll
        for (int r = 0; r < 8; ++r) acc[r] *= rscale[mh + r];
#if GAT_USE_ASYNC
        gat_wait_async();
#endif
        __syncthreads();                             // B3: p_lds/ht/red ready

        // ---- phase D: owners fold row sums; all waves do fragments+WMMA ----
        if (t < 16) {
            float s = 0.0f;
            #pragma unroll
            for (int s2 = 0; s2 < 16; ++s2) s += red[t][s2];
            rs[t] = rs[t] * rscale[t] + s;
        }
        v16bf a0, a1, b0, b1;
        #pragma unroll
        for (int e = 0; e < 16; ++e) {
            int ka = kbB + e;
            a0[e] = p_lds[m][ka];
            a1[e] = p_lds[m][32 + ka];
            b0[e] = ht[col][kbB + e];
            b1[e] = ht[col][32 + kbB + e];
        }
        acc = __builtin_amdgcn_wmma_f32_16x16x32_bf16(
            false, a0, false, b0, (short)0, acc, false, false);
        acc = __builtin_amdgcn_wmma_f32_16x16x32_bf16(
            false, a1, false, b1, (short)0, acc, false, false);
        __syncthreads();                             // B4: LDS free for next
    }

    // ---- finalize: normalize, ELU, store ----
    if (t < 16) rs[t] = 1.0f / rs[t];                // diag -> rs > 0
    __syncthreads();
    #pragma unroll
    for (int r = 0; r < 8; ++r) {
        float v = acc[r] * rs[mh + r];
        v = v > 0.0f ? v : (__expf(v) - 1.0f);
        out[(size_t)(row0 + mh + r) * GAT_FOUT + col] = v;
    }
}

// ---------------------------------------------------------------------------
extern "C" void kernel_launch(void* const* d_in, const int* in_sizes, int n_in,
                              void* d_out, int out_size, void* d_ws, size_t ws_size,
                              hipStream_t stream) {
    const float* x   = (const float*)d_in[0];   // [8192,512]
    const float* adj = (const float*)d_in[1];   // [8192,8192]
    const float* W   = (const float*)d_in[2];   // [512,128]
    const float* a   = (const float*)d_in[3];   // [256,1]
    float* out = (float*)d_out;                 // [8192,128]

    char* ws = (char*)d_ws;
    const size_t HBYTES = (size_t)GAT_N * GAT_FOUT * sizeof(__bf16);  // 2MB
    __bf16* h_bf  = (__bf16*)ws;                 // row-major [N][F_OUT]
    __bf16* h_t   = (__bf16*)(ws + HBYTES);      // transposed [F_OUT][N]
    float*  s_src = (float*)(ws + 2 * HBYTES);
    float*  s_dst = s_src + GAT_N;

    gat_k1_gemm_hW<<<GAT_N / 16, 256, 0, stream>>>(x, W, h_bf, h_t);
    gat_k2_logits<<<GAT_N, 128, 0, stream>>>(h_bf, a, s_src, s_dst);
    gat_k3_fused_attn<<<GAT_N / 16, 256, 0, stream>>>(adj, h_t, s_src, s_dst, out);
}
